// MoEMambaBlock_63015760167024
// MI455X (gfx1250) — compile-verified, matrix operand order checked by reading, not run on hardware
//
#include <hip/hip_runtime.h>
#include <hip/hip_bf16.h>

// ---------------- problem constants ----------------
constexpr int D_MODEL = 512;
constexpr int D_STATE = 16;
constexpr int D_TRANK = 32;            // (512+15)/16
constexpr int D_CONV  = 4;
constexpr int N_EXP   = 4;
constexpr int D_IN    = 2048;          // DIM*EXPAND
constexpr int N_HID   = 2048;
constexpr int N_BATCH = 2;
constexpr int N_SEQ   = 2048;
constexpr int N_TOK   = N_BATCH * N_SEQ;   // 4096
constexpr int N_DEPTH = 2;
constexpr int DBC_W   = D_TRANK + 2 * D_STATE;  // 64
constexpr float F_EPS = 1e-6f;

// ---------------- types ----------------
typedef __attribute__((ext_vector_type(16))) __bf16 v16bf;
typedef __attribute__((ext_vector_type(8)))  float  v8f;

struct alignas(16) U128 { unsigned int w[4]; };
union FragAB { v16bf v; U128 u[2]; };

__device__ __forceinline__ unsigned short f32_to_bf16_rne(float f) {
    unsigned int u = __float_as_uint(f);
    unsigned int lsb = (u >> 16) & 1u;
    u += 0x7fffu + lsb;                 // round-to-nearest-even
    return (unsigned short)(u >> 16);
}

// ---------------- generic NT GEMM: C[M,N] = A[M,K] * B[N,K]^T (+bias[N]) ----------------
// bf16 inputs (raw u16 storage), f32 accumulate via v_wmma_f32_16x16x32_bf16.
// Block = 128 threads = 4 waves in a 2x2 grid; each wave computes a 64x32 output
// tile (4x2 WMMA tiles) -> 8 WMMA per 12 b128 loads (1.5 loads/wmma).
// Block tile: 128 (M) x 64 (N).
__global__ __launch_bounds__(128)
void k_gemm_bf16_nt(const unsigned short* __restrict__ A, int lda,
                    const unsigned short* __restrict__ B, int ldb,
                    float* __restrict__ C, int ldc,
                    int M, int N, int K,
                    const float* __restrict__ bias)
{
    const int lane = threadIdx.x & 31;
    const int wave = threadIdx.x >> 5;
    const int wm   = wave >> 1;         // 0..1
    const int wn   = wave & 1;          // 0..1
    const int half = lane >> 4;         // 0: K 0..7 / 16..23 ; 1: K 8..15 / 24..31
    const int r    = lane & 15;

    const int m0 = blockIdx.y * 128 + wm * 64;
    const int n0 = blockIdx.x * 64  + wn * 32;
    if (m0 >= M || n0 >= N) return;     // wave-uniform: EXEC stays all-ones

    const unsigned short* ap0 = A + (size_t)(m0 + r)      * lda + half * 8;
    const unsigned short* ap1 = A + (size_t)(m0 + 16 + r) * lda + half * 8;
    const unsigned short* ap2 = A + (size_t)(m0 + 32 + r) * lda + half * 8;
    const unsigned short* ap3 = A + (size_t)(m0 + 48 + r) * lda + half * 8;
    const unsigned short* bp0 = B + (size_t)(n0 + r)      * ldb + half * 8;
    const unsigned short* bp1 = B + (size_t)(n0 + 16 + r) * ldb + half * 8;

    v8f c00 = {0.f,0.f,0.f,0.f,0.f,0.f,0.f,0.f};
    v8f c01 = c00, c10 = c00, c11 = c00;
    v8f c20 = c00, c21 = c00, c30 = c00, c31 = c00;

    for (int k0 = 0; k0 < K; k0 += 32) {
        FragAB a0, a1, a2, a3, b0, b1;
        a0.u[0] = *(const U128*)(ap0 + k0);
        a0.u[1] = *(const U128*)(ap0 + k0 + 16);
        a1.u[0] = *(const U128*)(ap1 + k0);
        a1.u[1] = *(const U128*)(ap1 + k0 + 16);
        a2.u[0] = *(const U128*)(ap2 + k0);
        a2.u[1] = *(const U128*)(ap2 + k0 + 16);
        a3.u[0] = *(const U128*)(ap3 + k0);
        a3.u[1] = *(const U128*)(ap3 + k0 + 16);
        b0.u[0] = *(const U128*)(bp0 + k0);
        b0.u[1] = *(const U128*)(bp0 + k0 + 16);
        b1.u[0] = *(const U128*)(bp1 + k0);
        b1.u[1] = *(const U128*)(bp1 + k0 + 16);
        __builtin_prefetch(ap0 + k0 + 128, 0, 3);   // global_prefetch_b8
        __builtin_prefetch(ap1 + k0 + 128, 0, 3);
        __builtin_prefetch(ap2 + k0 + 128, 0, 3);
        __builtin_prefetch(ap3 + k0 + 128, 0, 3);
        __builtin_prefetch(bp0 + k0 + 128, 0, 3);
        __builtin_prefetch(bp1 + k0 + 128, 0, 3);
        c00 = __builtin_amdgcn_wmma_f32_16x16x32_bf16(false, a0.v, false, b0.v, (short)0, c00, false, false);
        c01 = __builtin_amdgcn_wmma_f32_16x16x32_bf16(false, a0.v, false, b1.v, (short)0, c01, false, false);
        c10 = __builtin_amdgcn_wmma_f32_16x16x32_bf16(false, a1.v, false, b0.v, (short)0, c10, false, false);
        c11 = __builtin_amdgcn_wmma_f32_16x16x32_bf16(false, a1.v, false, b1.v, (short)0, c11, false, false);
        c20 = __builtin_amdgcn_wmma_f32_16x16x32_bf16(false, a2.v, false, b0.v, (short)0, c20, false, false);
        c21 = __builtin_amdgcn_wmma_f32_16x16x32_bf16(false, a2.v, false, b1.v, (short)0, c21, false, false);
        c30 = __builtin_amdgcn_wmma_f32_16x16x32_bf16(false, a3.v, false, b0.v, (short)0, c30, false, false);
        c31 = __builtin_amdgcn_wmma_f32_16x16x32_bf16(false, a3.v, false, b1.v, (short)0, c31, false, false);
    }

    // D layout: lanes 0..15 -> M = vgpr idx, N = lane; lanes 16..31 -> M = 8+idx
    auto store_tile = [&](const v8f& acc, int mt, int nt) {
        const int ncol  = nt + r;
        const int mbase = mt + half * 8;
        const float bv = bias ? bias[ncol] : 0.0f;
        #pragma unroll
        for (int i = 0; i < 8; ++i)
            C[(size_t)(mbase + i) * ldc + ncol] = acc[i] + bv;
    };
    store_tile(c00, m0,      n0);
    store_tile(c01, m0,      n0 + 16);
    store_tile(c10, m0 + 16, n0);
    store_tile(c11, m0 + 16, n0 + 16);
    store_tile(c20, m0 + 32, n0);
    store_tile(c21, m0 + 32, n0 + 16);
    store_tile(c30, m0 + 48, n0);
    store_tile(c31, m0 + 48, n0 + 16);
}

// ---------------- element-wise / small kernels ----------------
__global__ void k_f32_to_bf16(const float* __restrict__ in,
                              unsigned short* __restrict__ out, long n) {
    long i = (long)blockIdx.x * blockDim.x + threadIdx.x;
    if (i < n) out[i] = f32_to_bf16_rne(in[i]);
}

__global__ void k_moe_gate(const float* __restrict__ x,
                           const float* __restrict__ gw,
                           const float* __restrict__ gb,
                           float* __restrict__ masked) {
    int tok = blockIdx.x * blockDim.x + threadIdx.x;
    if (tok >= N_TOK) return;
    const float* xr = x + (size_t)tok * D_MODEL;
    float lg[N_EXP];
    for (int e = 0; e < N_EXP; ++e) {
        const float* w = gw + (size_t)e * D_MODEL;
        float acc = 0.f;
        for (int k = 0; k < D_MODEL; ++k) acc += xr[k] * w[k];
        lg[e] = acc + gb[e];
    }
    float mx = lg[0]; int am = 0;
    for (int e = 1; e < N_EXP; ++e) if (lg[e] > mx) { mx = lg[e]; am = e; }
    float p[N_EXP], den = 0.f;
    for (int e = 0; e < N_EXP; ++e) { p[e] = __expf(lg[e] - mx); den += p[e]; }
    for (int e = 0; e < N_EXP; ++e)
        masked[(size_t)tok * N_EXP + e] = (e == am) ? (p[e] / den) : 0.f;
}

__global__ void k_gate_norm(const float* __restrict__ masked,
                            float* __restrict__ gate) {
    int i = blockIdx.x * blockDim.x + threadIdx.x;   // over SEQ*N_EXP
    if (i >= N_SEQ * N_EXP) return;
    float m0 = masked[i];
    float m1 = masked[(size_t)N_SEQ * N_EXP + i];
    float inv = (float)N_BATCH / (m0 + m1 + F_EPS);  // capacity = int(1.0*BATCH)
    gate[i] = m0 * inv;
    gate[(size_t)N_SEQ * N_EXP + i] = m1 * inv;
}

__global__ void k_gelu_to_bf16(const float* __restrict__ in,
                               unsigned short* __restrict__ out, long n) {
    long i = (long)blockIdx.x * blockDim.x + threadIdx.x;
    if (i < n) {
        float v = in[i];
        float g = 0.5f * v * (1.0f + erff(v * 0.70710678118654752440f));
        out[i] = f32_to_bf16_rne(g);
    }
}

__global__ void k_moe_combine(const float* __restrict__ eo,
                              const float* __restrict__ gate,
                              float* __restrict__ acc, int e) {
    long i = (long)blockIdx.x * blockDim.x + threadIdx.x;
    if (i >= (long)N_TOK * D_MODEL) return;
    long tok = i / D_MODEL;
    float v = eo[i] * gate[tok * N_EXP + e];
    acc[i] = (e == 0) ? v : (acc[i] + v);
}

__global__ void k_conv_silu(const float* __restrict__ xz,
                            const float* __restrict__ cw,
                            const float* __restrict__ cb,
                            float* __restrict__ u,
                            unsigned short* __restrict__ ubf) {
    long i = (long)blockIdx.x * blockDim.x + threadIdx.x;   // over TOK*D_IN
    if (i >= (long)N_TOK * D_IN) return;
    int  c  = (int)(i % D_IN);
    long bl = i / D_IN;
    int  l  = (int)(bl % N_SEQ);
    int  b  = (int)(bl / N_SEQ);
    float acc = cb[c];
    const float* w = cw + (size_t)c * D_CONV;
    #pragma unroll
    for (int j = 0; j < D_CONV; ++j) {
        int t = l - (D_CONV - 1) + j;
        if (t >= 0)
            acc += xz[((size_t)(b * N_SEQ + t)) * (2 * D_IN) + c] * w[j];
    }
    float s = acc / (1.0f + __expf(-acc));     // silu
    u[i] = s;
    ubf[i] = f32_to_bf16_rne(s);
}

__global__ void k_softplus(float* __restrict__ d, long n) {
    long i = (long)blockIdx.x * blockDim.x + threadIdx.x;
    if (i < n) {
        float v = d[i];
        d[i] = (v > 20.f) ? v : log1pf(__expf(v));
    }
}

// Selective scan: one thread per (batch, channel). 16 states in registers.
__global__ void k_scan(const float* __restrict__ delta,
                       const float* __restrict__ dbc,
                       const float* __restrict__ u,
                       const float* __restrict__ A_log,
                       float* __restrict__ ys) {
    int id = blockIdx.x * blockDim.x + threadIdx.x;
    if (id >= N_BATCH * D_IN) return;
    int b = id / D_IN, d = id % D_IN;
    float A[D_STATE], s[D_STATE];
    #pragma unroll
    for (int n = 0; n < D_STATE; ++n) {
        A[n] = -__expf(A_log[(size_t)d * D_STATE + n]);
        s[n] = 0.f;
    }
    for (int t = 0; t < N_SEQ; ++t) {
        size_t row = (size_t)(b * N_SEQ + t);
        float dt = delta[row * D_IN + d];
        float ut = u[row * D_IN + d];
        float du = dt * ut;
        const float* bc = dbc + row * DBC_W;
        float y = 0.f;
        #pragma unroll
        for (int n = 0; n < D_STATE; ++n) {
            s[n] = __expf(dt * A[n]) * s[n] + du * bc[D_TRANK + n];
            y += s[n] * bc[D_TRANK + D_STATE + n];
        }
        ys[row * D_IN + d] = y;
    }
}

__global__ void k_ycombine(const float* __restrict__ ys,
                           const float* __restrict__ u,
                           const float* __restrict__ xz,
                           const float* __restrict__ Dp,
                           unsigned short* __restrict__ ybf) {
    long i = (long)blockIdx.x * blockDim.x + threadIdx.x;   // over TOK*D_IN
    if (i >= (long)N_TOK * D_IN) return;
    int  d   = (int)(i % D_IN);
    long row = i / D_IN;
    float res = xz[row * (2 * D_IN) + D_IN + d];
    float y = (ys[i] + u[i] * Dp[d]) * (res / (1.0f + __expf(-res)));
    ybf[i] = f32_to_bf16_rne(y);
}

__global__ void k_add(const float* __restrict__ a, const float* __restrict__ b,
                      float* __restrict__ out, long n) {
    long i = (long)blockIdx.x * blockDim.x + threadIdx.x;
    if (i < n) out[i] = a[i] + b[i];
}

// ---------------- host orchestration ----------------
extern "C" void kernel_launch(void* const* d_in, const int* in_sizes, int n_in,
                              void* d_out, int out_size, void* d_ws, size_t ws_size,
                              hipStream_t stream)
{
    const float* x_in   = (const float*)d_in[0];
    const float* in_w   = (const float*)d_in[1];
    const float* conv_w = (const float*)d_in[2];
    const float* conv_b = (const float*)d_in[3];
    const float* xprojw = (const float*)d_in[4];
    const float* dt_w   = (const float*)d_in[5];
    const float* dt_b   = (const float*)d_in[6];
    const float* A_log  = (const float*)d_in[7];
    const float* Dp     = (const float*)d_in[8];
    const float* out_w  = (const float*)d_in[9];
    const float* gate_w = (const float*)d_in[10];
    const float* gate_b = (const float*)d_in[11];
    const float* w1     = (const float*)d_in[12];
    const float* b1     = (const float*)d_in[13];
    const float* w2     = (const float*)d_in[14];
    const float* b2     = (const float*)d_in[15];

    char* ws = (char*)d_ws;
    size_t off = 0;
    auto alloc = [&](size_t bytes) -> void* {
        void* p = ws + off;
        off += (bytes + 255) & ~(size_t)255;
        return p;
    };

    typedef unsigned short bf16u;
    float* xa      = (float*)alloc((size_t)N_TOK * D_MODEL * 4);
    float* xb      = (float*)alloc((size_t)N_TOK * D_MODEL * 4);
    bf16u* xbf     = (bf16u*)alloc((size_t)N_TOK * D_MODEL * 2);
    bf16u* inw_bf  = (bf16u*)alloc((size_t)N_DEPTH * 2 * D_IN * D_MODEL * 2);
    bf16u* xprj_bf = (bf16u*)alloc((size_t)N_DEPTH * DBC_W * D_IN * 2);
    bf16u* dtw_bf  = (bf16u*)alloc((size_t)N_DEPTH * D_IN * D_TRANK * 2);
    bf16u* outw_bf = (bf16u*)alloc((size_t)N_DEPTH * D_MODEL * D_IN * 2);
    bf16u* w1_bf   = (bf16u*)alloc((size_t)N_DEPTH * N_EXP * N_HID * D_MODEL * 2);
    bf16u* w2_bf   = (bf16u*)alloc((size_t)N_DEPTH * N_EXP * D_MODEL * N_HID * 2);
    float* masked  = (float*)alloc((size_t)N_TOK * N_EXP * 4);
    float* gate    = (float*)alloc((size_t)N_TOK * N_EXP * 4);
    float* xz      = (float*)alloc((size_t)N_TOK * 2 * D_IN * 4);   // mamba; MoE aliases h in here
    float* u_f     = (float*)alloc((size_t)N_TOK * D_IN * 4);       // MoE aliases eo in here
    bf16u* ubf     = (bf16u*)alloc((size_t)N_TOK * D_IN * 2);
    float* dbc     = (float*)alloc((size_t)N_TOK * DBC_W * 4);
    bf16u* dbcbf   = (bf16u*)alloc((size_t)N_TOK * DBC_W * 2);
    float* delta   = (float*)alloc((size_t)N_TOK * D_IN * 4);
    float* ys      = (float*)alloc((size_t)N_TOK * D_IN * 4);
    bf16u* ybf     = (bf16u*)alloc((size_t)N_TOK * D_IN * 2);
    float* mout    = (float*)alloc((size_t)N_TOK * D_MODEL * 4);

    // phase aliases (MoE and Mamba never overlap in time)
    float* hpre = xz;                                        // 32MB of 64MB region
    bf16u* hbf  = (bf16u*)(xz + (size_t)N_TOK * N_HID);      // next 16MB
    float* eo   = u_f;

    const int THR = 256;
    auto cvt = [&](const float* src, bf16u* dst, long n) {
        k_f32_to_bf16<<<(unsigned)((n + THR - 1) / THR), THR, 0, stream>>>(src, dst, n);
    };
    auto gemm = [&](const bf16u* A, int lda, const bf16u* B, int ldb,
                    float* C, int ldc, int M, int N, int K, const float* bias) {
        dim3 grid((unsigned)(N / 64), (unsigned)(M / 128));
        k_gemm_bf16_nt<<<grid, 128, 0, stream>>>(A, lda, B, ldb, C, ldc, M, N, K, bias);
    };

    // --- weight conversion to bf16 (re-done every launch: deterministic) ---
    cvt(in_w,   inw_bf,  (long)N_DEPTH * 2 * D_IN * D_MODEL);
    cvt(xprojw, xprj_bf, (long)N_DEPTH * DBC_W * D_IN);
    cvt(dt_w,   dtw_bf,  (long)N_DEPTH * D_IN * D_TRANK);
    cvt(out_w,  outw_bf, (long)N_DEPTH * D_MODEL * D_IN);
    cvt(w1,     w1_bf,   (long)N_DEPTH * N_EXP * N_HID * D_MODEL);
    cvt(w2,     w2_bf,   (long)N_DEPTH * N_EXP * D_MODEL * N_HID);

    auto run_moe = [&](const float* xin, float* xout, int L) {
        cvt(xin, xbf, (long)N_TOK * D_MODEL);
        k_moe_gate<<<(N_TOK + 127) / 128, 128, 0, stream>>>(
            xin, gate_w + (size_t)L * N_EXP * D_MODEL, gate_b + (size_t)L * N_EXP, masked);
        k_gate_norm<<<(N_SEQ * N_EXP + THR - 1) / THR, THR, 0, stream>>>(masked, gate);
        for (int e = 0; e < N_EXP; ++e) {
            const bf16u* w1e = w1_bf + ((size_t)L * N_EXP + e) * N_HID * D_MODEL;
            const bf16u* w2e = w2_bf + ((size_t)L * N_EXP + e) * D_MODEL * N_HID;
            gemm(xbf, D_MODEL, w1e, D_MODEL, hpre, N_HID,
                 N_TOK, N_HID, D_MODEL, b1 + ((size_t)L * N_EXP + e) * N_HID);
            long nh = (long)N_TOK * N_HID;
            k_gelu_to_bf16<<<(unsigned)((nh + THR - 1) / THR), THR, 0, stream>>>(hpre, hbf, nh);
            gemm(hbf, N_HID, w2e, N_HID, eo, D_MODEL,
                 N_TOK, D_MODEL, N_HID, b2 + ((size_t)L * N_EXP + e) * D_MODEL);
            long nd = (long)N_TOK * D_MODEL;
            k_moe_combine<<<(unsigned)((nd + THR - 1) / THR), THR, 0, stream>>>(eo, gate, xout, e);
        }
    };

    auto run_mamba = [&](const float* xin, int L) {
        cvt(xin, xbf, (long)N_TOK * D_MODEL);
        gemm(xbf, D_MODEL, inw_bf + (size_t)L * 2 * D_IN * D_MODEL, D_MODEL,
             xz, 2 * D_IN, N_TOK, 2 * D_IN, D_MODEL, nullptr);
        long nu = (long)N_TOK * D_IN;
        k_conv_silu<<<(unsigned)((nu + THR - 1) / THR), THR, 0, stream>>>(
            xz, conv_w + (size_t)L * D_IN * D_CONV, conv_b + (size_t)L * D_IN, u_f, ubf);
        gemm(ubf, D_IN, xprj_bf + (size_t)L * DBC_W * D_IN, D_IN,
             dbc, DBC_W, N_TOK, DBC_W, D_IN, nullptr);
        cvt(dbc, dbcbf, (long)N_TOK * DBC_W);
        gemm(dbcbf, DBC_W, dtw_bf + (size_t)L * D_IN * D_TRANK, D_TRANK,
             delta, D_IN, N_TOK, D_IN, D_TRANK, dt_b + (size_t)L * D_IN);
        k_softplus<<<(unsigned)((nu + THR - 1) / THR), THR, 0, stream>>>(delta, nu);
        k_scan<<<(N_BATCH * D_IN + 127) / 128, 128, 0, stream>>>(
            delta, dbc, u_f, A_log + (size_t)L * D_IN * D_STATE, ys);
        k_ycombine<<<(unsigned)((nu + THR - 1) / THR), THR, 0, stream>>>(
            ys, u_f, xz, Dp + (size_t)L * D_IN, ybf);
        gemm(ybf, D_IN, outw_bf + (size_t)L * D_MODEL * D_IN, D_IN,
             mout, D_MODEL, N_TOK, D_MODEL, D_IN, nullptr);
    };

    const float* xcur = x_in;
    long nx = (long)N_TOK * D_MODEL;
    for (int L = 0; L < N_DEPTH; ++L) {
        run_moe(xcur, xb, L);                 // x = moe(x)
        run_mamba(xb, L);                     // mamba(x) -> mout
        k_add<<<(unsigned)((nx + THR - 1) / THR), THR, 0, stream>>>(mout, xb, xa, nx);
        run_moe(xa, xb, L);                   // x = moe(x)
        xcur = xb;
    }
    hipMemcpyAsync(d_out, xcur, (size_t)N_TOK * D_MODEL * sizeof(float),
                   hipMemcpyDeviceToDevice, stream);
}
